// SingleViewGraphEncoder_39900246180619
// MI455X (gfx1250) — matrix operand order repaired and checked by reading, not compiled, not run on hardware
//
#include <hip/hip_runtime.h>
#include <hip/hip_bf16.h>
#include <math.h>

typedef __bf16 bf16_t;
typedef __attribute__((ext_vector_type(16))) __bf16 v16bf;
typedef __attribute__((ext_vector_type(8)))  float  v8f;

#define NNODES 6144
#define DIN    512
#define HDIM   512
#define ODIM   256
#define EPSBN  1e-5f

// POD 16-byte chunk (HIP's uint4 has ctors -> not union-safe)
struct alignas(16) U128 { unsigned int a, b, c, d; };

// ---- CDNA5 async global->LDS staging (ASYNCcnt), with sync fallback ----
#if __has_builtin(__builtin_amdgcn_global_load_async_to_lds_b128) && \
    __has_builtin(__builtin_amdgcn_s_wait_asynccnt)
#define HAVE_ASYNC_LDS 1
#else
#define HAVE_ASYNC_LDS 0
#endif

#if HAVE_ASYNC_LDS
// builtin signature (probe-confirmed): pointers to 'int __attribute__((vector_size(16)))'
// in AS1 (global) and AS3 (LDS)
typedef int v4si __attribute__((vector_size(16)));
typedef __attribute__((address_space(1))) v4si as1_v4si;
typedef __attribute__((address_space(3))) v4si as3_v4si;
#endif

__device__ __forceinline__ void g2l_b128(const void* g, void* l) {
#if HAVE_ASYNC_LDS
  // flat global address == AS1 address; low 32 bits of a flat LDS address == LDS offset
  __builtin_amdgcn_global_load_async_to_lds_b128(
      (as1_v4si*)(unsigned long long)(size_t)g,
      (as3_v4si*)(unsigned)(size_t)l, 0, 0);
#else
  *(U128*)l = *(const U128*)g;
#endif
}

template <unsigned short NPEND>
__device__ __forceinline__ void wait_async() {
#if HAVE_ASYNC_LDS
  __builtin_amdgcn_s_wait_asynccnt(NPEND);
#endif
}

// ---------------- elementwise / reduction kernels ----------------

__global__ __launch_bounds__(256)
void k_rowsum_rsqrt(const float* __restrict__ A, float* __restrict__ dis, int n) {
  __shared__ float red[256];
  const int row = blockIdx.x;
  const float* p = A + (size_t)row * n;
  float s = 0.f;
  for (int c = threadIdx.x; c < n; c += 256) s += p[c];
  red[threadIdx.x] = s; __syncthreads();
  for (int o = 128; o > 0; o >>= 1) {
    if (threadIdx.x < o) red[threadIdx.x] += red[threadIdx.x + o];
    __syncthreads();
  }
  if (threadIdx.x == 0) {
    float d = red[0];
    dis[row] = (d > 0.f) ? rsqrtf(d) : 0.f;
  }
}

__global__ __launch_bounds__(256)
void k_cvt_bf16(const float* __restrict__ in, bf16_t* __restrict__ out, size_t n) {
  size_t i = (size_t)blockIdx.x * 256 + threadIdx.x;
  if (i < n) out[i] = (bf16_t)in[i];
}

// outT[c][j] = X[j][c] * s[j]   (X: R x D row-major; outT: D x R row-major)
// Output-centric indexing: bf16 stores fully coalesced; one-off strided f32 reads.
__global__ __launch_bounds__(256)
void k_scale_rows_bf16_T(const float* __restrict__ X, const float* __restrict__ s,
                         bf16_t* __restrict__ outT, int D, int R) {
  size_t i = (size_t)blockIdx.x * 256 + threadIdx.x;
  if (i >= (size_t)D * R) return;
  int c = (int)(i / (size_t)R);
  int j = (int)(i % (size_t)R);
  outT[i] = (bf16_t)(X[(size_t)j * D + c] * s[j]);
}

// ---------------- bf16 WMMA GEMM (double-buffered, fully async staging) ----------------
// C[M,N] = alpha * A[M,K] @ B^T,  A row-major MxK, B row-major NxK (both K-contiguous).
// Block tile 128x128, BK=32; 8 wave32 waves, each wave owns 32x64 (8 WMMAs/K-step).
// Epilogue (compile-time flags): rowscale[m], bias[n], adjacency mask (0 -> -inf),
// f32 out, bf16 out, or bf16 *transposed* out (vectorized b128 column stores).
// Requires M%128==0, N%128==0, K%32==0 (true for every call here).
template <bool BIAS, bool RS, bool MASK, bool OUTF, bool OUTB, bool OUTBT>
__global__ __launch_bounds__(256)
void k_gemm_bf16(const bf16_t* __restrict__ A, const bf16_t* __restrict__ B,
                 int M, int N, int K, float alpha,
                 const float* __restrict__ bias,
                 const float* __restrict__ rowscale,
                 const float* __restrict__ mask,
                 float* __restrict__ outF, bf16_t* __restrict__ outB) {
  __shared__ __align__(64) bf16_t lA[2][128][32];  // [m][k]
  __shared__ __align__(64) bf16_t lB[2][128][32];  // [n][k]

  const int tid  = threadIdx.x;
  const int lane = tid & 31;
  const int wave = tid >> 5;
  const int m0 = blockIdx.y * 128;
  const int n0 = blockIdx.x * 128;
  const int wm = (wave & 3) * 32;   // wave's 32 rows
  const int wn = (wave >> 2) * 64;  // wave's 64 cols

  v8f acc[2][4];
#pragma unroll
  for (int i = 0; i < 2; ++i)
#pragma unroll
    for (int j = 0; j < 4; ++j)
#pragma unroll
      for (int r = 0; r < 8; ++r) acc[i][j][r] = 0.f;

  constexpr unsigned short PER_TILE = 4;  // async b128 ops per thread per tile

  const int sRow = tid >> 2;          // 0..63
  const int sCk  = (tid & 3) * 8;
  auto stage = [&](int buf, int k0) {
    // A tile: 128x32 bf16 = 512 x b128; 2 chunks/thread (async)
    g2l_b128(A + (size_t)(m0 + sRow) * K + k0 + sCk,       &lA[buf][sRow][sCk]);
    g2l_b128(A + (size_t)(m0 + sRow + 64) * K + k0 + sCk,  &lA[buf][sRow + 64][sCk]);
    // B tile: 128x32 bf16, rows K-contiguous (async)
    g2l_b128(B + (size_t)(n0 + sRow) * K + k0 + sCk,       &lB[buf][sRow][sCk]);
    g2l_b128(B + (size_t)(n0 + sRow + 64) * K + k0 + sCk,  &lB[buf][sRow + 64][sCk]);
  };

  // per-lane fragment geometry (wave32; k = 16*(lane/16) + s for A and B)
  const int frow  = lane & 15;            // A row / B col within 16-tile
  const int abase = (lane < 16) ? 0 : 16; // A half-wave K base
  const int kb    = (lane < 16) ? 0 : 16; // B half-wave K base

  auto compute = [&](int cur) {
    union AFrag { U128 u[2]; v16bf v; } af[2];
#pragma unroll
    for (int i = 0; i < 2; ++i) {
      af[i].u[0] = *(const U128*)&lA[cur][wm + i * 16 + frow][abase];
      af[i].u[1] = *(const U128*)&lA[cur][wm + i * 16 + frow][abase + 8];
    }
    v16bf bfr[4];
#pragma unroll
    for (int j = 0; j < 4; ++j)
      bfr[j] = *(const v16bf*)&lB[cur][wn + j * 16 + frow][kb];
#pragma unroll
    for (int i = 0; i < 2; ++i)
#pragma unroll
      for (int j = 0; j < 4; ++j)
        acc[i][j] = __builtin_amdgcn_wmma_f32_16x16x32_bf16(
            false, af[i].v, false, bfr[j], (short)0, acc[i][j], false, false);
  };

  const int nt = K / 32;  // >= 16 for all calls
  stage(0, 0);
  for (int t = 0; t < nt - 1; ++t) {
    const int cur = t & 1;
    stage(cur ^ 1, (t + 1) * 32);  // prefetch next tile into other buffer
    wait_async<PER_TILE>();        // tile t landed; tile t+1 still in flight
    __syncthreads();
    compute(cur);
    __syncthreads();
  }
  wait_async<0>();
  __syncthreads();
  compute((nt - 1) & 1);

  // ---- templated epilogue ----
  const int colL = lane & 15;
  const int rH   = (lane >> 4) * 8;
#pragma unroll
  for (int i = 0; i < 2; ++i)
#pragma unroll
    for (int j = 0; j < 4; ++j) {
      const int nc = n0 + wn + j * 16 + colL;
      const int mb = m0 + wm + i * 16 + rH;   // 8 contiguous M rows start here
      const float bv = BIAS ? bias[nc] : 0.f;
      if constexpr (OUTBT) {
        // transposed bf16 store: 8 contiguous rows -> one b128 column store
        union { U128 u; bf16_t h[8]; } pk;
#pragma unroll
        for (int r = 0; r < 8; ++r) {
          float v = acc[i][j][r] * alpha;
          if constexpr (RS)   v *= rowscale[mb + r];
          if constexpr (BIAS) v += bv;
          pk.h[r] = (bf16_t)v;
        }
        *(U128*)(outB + (size_t)nc * M + mb) = pk.u;
      } else {
#pragma unroll
        for (int r = 0; r < 8; ++r) {
          const int mm = mb + r;
          float v = acc[i][j][r] * alpha;
          if constexpr (RS)   v *= rowscale[mm];
          if constexpr (BIAS) v += bv;
          if constexpr (MASK) { if (mask[(size_t)mm * N + nc] == 0.f) v = -INFINITY; }
          const size_t idx = (size_t)mm * N + nc;
          if constexpr (OUTF) outF[idx] = v;
          if constexpr (OUTB) outB[idx] = (bf16_t)v;
        }
      }
    }
}

// ---------------- masked row softmax: f32 scores -> bf16 probs ----------------
__global__ __launch_bounds__(256)
void k_masked_softmax(const float* __restrict__ S, bf16_t* __restrict__ P, int n) {
  __shared__ float red[256];
  const int row = blockIdx.x;
  const float* s = S + (size_t)row * n;

  float mx = -INFINITY;
  for (int c = threadIdx.x; c < n; c += 256) mx = fmaxf(mx, s[c]);
  red[threadIdx.x] = mx; __syncthreads();
  for (int o = 128; o > 0; o >>= 1) {
    if (threadIdx.x < o) red[threadIdx.x] = fmaxf(red[threadIdx.x], red[threadIdx.x + o]);
    __syncthreads();
  }
  mx = red[0]; __syncthreads();

  float sum = 0.f;
  for (int c = threadIdx.x; c < n; c += 256) {
    float v = s[c];
    sum += (v == -INFINITY) ? 0.f : __expf(v - mx);
  }
  red[threadIdx.x] = sum; __syncthreads();
  for (int o = 128; o > 0; o >>= 1) {
    if (threadIdx.x < o) red[threadIdx.x] += red[threadIdx.x + o];
    __syncthreads();
  }
  const float inv = 1.f / red[0];

  bf16_t* p = P + (size_t)row * n;
  for (int c = threadIdx.x; c < n; c += 256) {
    float v = s[c];
    float e = (v == -INFINITY) ? 0.f : __expf(v - mx);
    p[c] = (bf16_t)(e * inv);
  }
}

// ---------------- BatchNorm (training-mode batch stats) ----------------
__global__ __launch_bounds__(256)
void k_bn_stats(const float* __restrict__ X, int rows, int C,
                float* __restrict__ mean, float* __restrict__ var) {
  int j = blockIdx.x * 256 + threadIdx.x;
  if (j >= C) return;
  float s = 0.f, s2 = 0.f;
  for (int i = 0; i < rows; ++i) {
    float v = X[(size_t)i * C + j];
    s += v; s2 += v * v;
  }
  float m = s / rows;
  mean[j] = m;
  var[j]  = s2 / rows - m * m;   // biased variance (matches jnp.var)
}

__global__ __launch_bounds__(256)
void k_bn_apply(const float* __restrict__ X, const float* __restrict__ mean,
                const float* __restrict__ var, const float* __restrict__ g,
                const float* __restrict__ beta, const float* __restrict__ resid,
                int relu, int C, size_t total,
                float* __restrict__ outF, bf16_t* __restrict__ outB,
                float* __restrict__ outF2) {
  size_t i = (size_t)blockIdx.x * 256 + threadIdx.x;
  if (i >= total) return;
  int j = (int)(i % (size_t)C);
  float v = g[j] * (X[i] - mean[j]) * rsqrtf(var[j] + EPSBN) + beta[j];
  if (relu)  v = fmaxf(v, 0.f);
  if (resid) v += resid[i];
  if (outF)  outF[i]  = v;
  if (outB)  outB[i]  = (bf16_t)v;
  if (outF2) outF2[i] = v;
}

// ---------------- host orchestration ----------------

extern "C" void kernel_launch(void* const* d_in, const int* in_sizes, int n_in,
                              void* d_out, int out_size, void* d_ws, size_t ws_size,
                              hipStream_t stream) {
  (void)in_sizes; (void)n_in; (void)out_size; (void)ws_size;

  const float* X        = (const float*)d_in[0];
  const float* adj      = (const float*)d_in[1];
  const float* adj_glo  = (const float*)d_in[2];
  const float* Wp       = (const float*)d_in[3];
  const float* bp       = (const float*)d_in[4];
  const float* gp       = (const float*)d_in[5];
  const float* betap    = (const float*)d_in[6];

  char* ws = (char*)d_ws;
  size_t off = 0;
  auto alloc = [&](size_t bytes) -> void* {
    void* p = ws + off;
    off = (off + bytes + 255) & ~(size_t)255;
    return p;
  };

  const size_t NN = (size_t)NNODES * NNODES;
  const size_t NH = (size_t)NNODES * HDIM;

  float*  dis  = (float*)alloc((size_t)NNODES * 4);
  float*  S    = (float*)alloc(NN * 4);            // f32 scores
  bf16_t* Gb   = (bf16_t*)alloc(NN * 2);           // adj_glo bf16, reused as attn probs
  bf16_t* XsT  = (bf16_t*)alloc((size_t)NNODES * DIN * 2);  // (dis_j * X)^T : DIN x N
  bf16_t* hb   = (bf16_t*)alloc(NH * 2);
  bf16_t* Qb   = (bf16_t*)alloc(NH * 2);
  bf16_t* Kb   = (bf16_t*)alloc(NH * 2);
  bf16_t* Vt   = (bf16_t*)alloc(NH * 2);           // V^T : HDIM x N
  bf16_t* Tb   = (bf16_t*)alloc(NH * 2);
  float*  h    = (float*)alloc(NH * 4);
  float*  U    = (float*)alloc(NH * 4);
  float*  mean = (float*)alloc(HDIM * 4);
  float*  var  = (float*)alloc(HDIM * 4);
  bf16_t* Wpb  = (bf16_t*)alloc((size_t)HDIM * DIN * 2);
  bf16_t *Wqb[3], *Wkb[3], *Wvb[3], *Wob[3];
  for (int i = 0; i < 3; ++i) {
    int o = (i < 2) ? HDIM : ODIM;
    Wqb[i] = (bf16_t*)alloc((size_t)HDIM * HDIM * 2);
    Wkb[i] = (bf16_t*)alloc((size_t)HDIM * HDIM * 2);
    Wvb[i] = (bf16_t*)alloc((size_t)HDIM * HDIM * 2);
    Wob[i] = (bf16_t*)alloc((size_t)o * HDIM * 2);
  }

  auto cvt = [&](const float* src, bf16_t* dst, size_t n) {
    k_cvt_bf16<<<dim3((unsigned)((n + 255) / 256)), dim3(256), 0, stream>>>(src, dst, n);
  };

  // --- setup: degrees, bf16 conversions ---
  k_rowsum_rsqrt<<<NNODES, 256, 0, stream>>>(adj_glo, dis, NNODES);
  cvt(adj_glo, Gb, NN);
  {
    size_t nX = (size_t)NNODES * DIN;
    k_scale_rows_bf16_T<<<dim3((unsigned)((nX + 255) / 256)), dim3(256), 0, stream>>>(
        X, dis, XsT, DIN, NNODES);
  }
  cvt(Wp, Wpb, (size_t)HDIM * DIN);
  for (int i = 0; i < 3; ++i) {
    int o = (i < 2) ? HDIM : ODIM;
    cvt((const float*)d_in[7 + i * 7 + 0], Wqb[i], (size_t)HDIM * HDIM);
    cvt((const float*)d_in[7 + i * 7 + 1], Wkb[i], (size_t)HDIM * HDIM);
    cvt((const float*)d_in[7 + i * 7 + 2], Wvb[i], (size_t)HDIM * HDIM);
    cvt((const float*)d_in[7 + i * 7 + 3], Wob[i], (size_t)o * HDIM);
  }

  // --- embedding: h0 = dis_i * (adj_glo @ Xs) = dis_i * (Gb @ XsT^T) -> bf16 in Qb ---
  k_gemm_bf16<false, true, false, false, true, false>
      <<<dim3(DIN / 128, NNODES / 128), dim3(256), 0, stream>>>(
          Gb, XsT, NNODES, DIN, NNODES, 1.f, nullptr, dis, nullptr, nullptr, Qb);
  // projection: U = h0 @ Wp^T + bp
  k_gemm_bf16<true, false, false, true, false, false>
      <<<dim3(HDIM / 128, NNODES / 128), dim3(256), 0, stream>>>(
          Qb, Wpb, NNODES, HDIM, DIN, 1.f, bp, nullptr, nullptr, U, nullptr);
  // BN (no relu); also writes embed_features slice of d_out
  k_bn_stats<<<dim3((HDIM + 255) / 256), dim3(256), 0, stream>>>(U, NNODES, HDIM, mean, var);
  k_bn_apply<<<dim3((unsigned)((NH + 255) / 256)), dim3(256), 0, stream>>>(
      U, mean, var, gp, betap, nullptr, 0, HDIM, NH,
      h, hb, (float*)d_out + (size_t)NNODES * ODIM);

  const float inv_sqrt_h = 0.044194173824159216f;  // 1/sqrt(512)

  for (int i = 0; i < 3; ++i) {
    const int o = (i < 2) ? HDIM : ODIM;
    const float* bo = (const float*)d_in[7 + i * 7 + 4];
    const float* g  = (const float*)d_in[7 + i * 7 + 5];
    const float* bt = (const float*)d_in[7 + i * 7 + 6];

    // Q, K normal bf16; V written transposed (Vt = V^T) for the async attn@V GEMM
    k_gemm_bf16<false, false, false, false, true, false>
        <<<dim3(HDIM / 128, NNODES / 128), dim3(256), 0, stream>>>(
            hb, Wqb[i], NNODES, HDIM, HDIM, 1.f, nullptr, nullptr, nullptr, nullptr, Qb);
    k_gemm_bf16<false, false, false, false, true, false>
        <<<dim3(HDIM / 128, NNODES / 128), dim3(256), 0, stream>>>(
            hb, Wkb[i], NNODES, HDIM, HDIM, 1.f, nullptr, nullptr, nullptr, nullptr, Kb);
    k_gemm_bf16<false, false, false, false, false, true>
        <<<dim3(HDIM / 128, NNODES / 128), dim3(256), 0, stream>>>(
            hb, Wvb[i], NNODES, HDIM, HDIM, 1.f, nullptr, nullptr, nullptr, nullptr, Vt);
    // scores = (Q @ K^T)/sqrt(H) with fused adjacency mask -> f32
    k_gemm_bf16<false, false, true, true, false, false>
        <<<dim3(NNODES / 128, NNODES / 128), dim3(256), 0, stream>>>(
            Qb, Kb, NNODES, NNODES, HDIM, inv_sqrt_h, nullptr, nullptr, adj, S, nullptr);
    // softmax rows -> bf16 probs (reuse Gb)
    k_masked_softmax<<<NNODES, 256, 0, stream>>>(S, Gb, NNODES);
    // T = attn @ V = attn @ Vt^T (bf16, fully async staging)
    k_gemm_bf16<false, false, false, false, true, false>
        <<<dim3(HDIM / 128, NNODES / 128), dim3(256), 0, stream>>>(
            Gb, Vt, NNODES, HDIM, NNODES, 1.f, nullptr, nullptr, nullptr, nullptr, Tb);
    // U = T @ Wo^T + bo (f32)
    k_gemm_bf16<true, false, false, true, false, false>
        <<<dim3(o / 128, NNODES / 128), dim3(256), 0, stream>>>(
            Tb, Wob[i], NNODES, o, HDIM, 1.f, bo, nullptr, nullptr, U, nullptr);
    // BN + ReLU (+ residual for layers 0,1); layer 2 writes d_out
    k_bn_stats<<<dim3((o + 255) / 256), dim3(256), 0, stream>>>(U, NNODES, o, mean, var);
    const size_t tot = (size_t)NNODES * o;
    k_bn_apply<<<dim3((unsigned)((tot + 255) / 256)), dim3(256), 0, stream>>>(
        U, mean, var, g, bt,
        (i < 2) ? h : nullptr, 1, o, tot,
        (i < 2) ? h : (float*)d_out,
        (i < 2) ? hb : nullptr,
        nullptr);
  }
}
